// RN_14843406975513
// MI455X (gfx1250) — matrix operand order repaired and checked by reading, not compiled, not run on hardware
//
#include <hip/hip_runtime.h>
#include <hip/hip_bf16.h>
#include <math.h>

// ---------------------------------------------------------------------------
// Types for CDNA5 WMMA (wave32, 16x16x32 bf16 -> f32 accum)
// ---------------------------------------------------------------------------
typedef __attribute__((ext_vector_type(16))) __bf16 v16bf;
typedef __attribute__((ext_vector_type(8)))  float  v8f;

union Frag16 {
    v16bf v;
    uint4 q[2];
};

static __device__ __forceinline__ unsigned short f2bf(float f) {
    unsigned int u = __float_as_uint(f);
    u += 0x7fffu + ((u >> 16) & 1u);   // round-to-nearest-even
    return (unsigned short)(u >> 16);
}

static __device__ __forceinline__ float sigmoidf_(float x) {
    return 1.0f / (1.0f + __expf(-x));
}

#define B_   64
#define T_   18
#define HID_ 128
#define EMB_ 32
#define GD_  256
#define AV_  28

// ---------------------------------------------------------------------------
// LSTM question encoder: one block per batch element, 128 threads (one per
// hidden unit). h lives in LDS; c in a register. Output = final CELL state.
// ---------------------------------------------------------------------------
__global__ void __launch_bounds__(128) lstm_k(
    const int* __restrict__ idx, const float* __restrict__ emb,
    const float* __restrict__ W_ih, const float* __restrict__ W_hh,
    const float* __restrict__ b_ih, const float* __restrict__ b_hh,
    const float* __restrict__ h0, const float* __restrict__ c0,
    float* __restrict__ qst)
{
    __shared__ float xs[EMB_];
    __shared__ float hs[HID_];
    const int b = blockIdx.x;
    const int j = threadIdx.x;
    float c = c0[b * HID_ + j];
    hs[j] = h0[b * HID_ + j];
    __syncthreads();
    for (int t = 0; t < T_; ++t) {
        const int w = idx[b * T_ + t];
        if (j < EMB_) xs[j] = (w != 0) ? emb[w * EMB_ + j] : 0.0f;
        __syncthreads();
        float g[4];
        #pragma unroll
        for (int gi = 0; gi < 4; ++gi) {
            const int row = gi * HID_ + j;
            float acc = b_ih[row] + b_hh[row];
            const float* wi = W_ih + row * EMB_;
            const float* wh = W_hh + row * HID_;
            for (int e = 0; e < EMB_; ++e) acc += xs[e] * wi[e];
            for (int e = 0; e < HID_; ++e) acc += hs[e] * wh[e];
            g[gi] = acc;
        }
        __syncthreads();
        const float ig = sigmoidf_(g[0]);
        const float fg = sigmoidf_(g[1]);
        const float gg = tanhf(g[2]);
        const float og = sigmoidf_(g[3]);
        c = fg * c + ig * gg;
        hs[j] = og * tanhf(c);
        __syncthreads();
    }
    qst[b * HID_ + j] = c;   // reference uses final cell state
}

// ---------------------------------------------------------------------------
// Direct conv (3x3, stride 2, pad 1) + bias + relu
// ---------------------------------------------------------------------------
__global__ void conv_relu_k(const float* __restrict__ x, const float* __restrict__ w,
                            const float* __restrict__ bias, float* __restrict__ y,
                            int Cin, int Cout, int Hin, int Win, int Hout, int Wout)
{
    const int total = B_ * Cout * Hout * Wout;
    int i = blockIdx.x * blockDim.x + threadIdx.x;
    if (i >= total) return;
    const int wo = i % Wout;
    const int ho = (i / Wout) % Hout;
    const int co = (i / (Wout * Hout)) % Cout;
    const int b  = i / (Wout * Hout * Cout);
    float acc = bias[co];
    for (int ci = 0; ci < Cin; ++ci) {
        const float* xp = x + ((size_t)(b * Cin + ci) * Hin) * Win;
        const float* wp = w + ((co * Cin + ci) * 9);
        #pragma unroll
        for (int kh = 0; kh < 3; ++kh) {
            const int hi = ho * 2 - 1 + kh;
            if (hi < 0 || hi >= Hin) continue;
            #pragma unroll
            for (int kw = 0; kw < 3; ++kw) {
                const int wi = wo * 2 - 1 + kw;
                if (wi < 0 || wi >= Win) continue;
                acc += xp[hi * Win + wi] * wp[kh * 3 + kw];
            }
        }
    }
    y[i] = fmaxf(acc, 0.0f);
}

// Train-mode BN stats: one block per channel computes sum and sumsq.
__global__ void __launch_bounds__(256) bn_stats_k(const float* __restrict__ y,
                                                  float* __restrict__ stats,
                                                  int C, int HW)
{
    __shared__ float s0[256], s1[256];
    const int c = blockIdx.x;
    const int t = threadIdx.x;
    const int N = B_ * HW;
    float a = 0.f, b = 0.f;
    for (int n = t; n < N; n += 256) {
        const int bb = n / HW, hw = n % HW;
        const float v = y[((size_t)(bb * C + c) * HW) + hw];
        a += v; b += v * v;
    }
    s0[t] = a; s1[t] = b;
    __syncthreads();
    for (int s = 128; s > 0; s >>= 1) {
        if (t < s) { s0[t] += s0[t + s]; s1[t] += s1[t + s]; }
        __syncthreads();
    }
    if (t == 0) { stats[2 * c] = s0[0]; stats[2 * c + 1] = s1[0]; }
}

__global__ void bn_apply_k(float* __restrict__ y, const float* __restrict__ stats,
                           const float* __restrict__ gamma, const float* __restrict__ beta,
                           int C, int HW)
{
    const int total = B_ * C * HW;
    int i = blockIdx.x * blockDim.x + threadIdx.x;
    if (i >= total) return;
    const int c = (i / HW) % C;
    const float N = (float)(B_ * HW);
    const float m = stats[2 * c] / N;
    const float v = stats[2 * c + 1] / N - m * m;
    y[i] = gamma[c] * (y[i] - m) * rsqrtf(v + 1e-5f) + beta[c];
}

// ---------------------------------------------------------------------------
// xf[b,p,0:24] = y4[b,ch,p]; xf[b,p,24:26] = coords(p)
// ---------------------------------------------------------------------------
__global__ void build_xf_k(const float* __restrict__ y4, float* __restrict__ xf)
{
    int i = blockIdx.x * blockDim.x + threadIdx.x;
    if (i >= B_ * 64 * 26) return;
    const int c = i % 26;
    const int p = (i / 26) % 64;
    const int b = i / (26 * 64);
    float v;
    if (c < 24)       v = y4[(b * 24 + c) * 64 + p];
    else if (c == 24) v = ((float)p / 8.0f - 4.0f) * 0.25f;   // python3 float div
    else              v = ((float)(p & 7) - 4.0f) * 0.25f;
    xf[i] = v;
}

// Ai = xf @ Wg1[:, :26].T ; Aj = xf @ Wg1[:, 26:52].T
__global__ void proj_a_k(const float* __restrict__ xf, const float* __restrict__ Wg1,
                         float* __restrict__ Ai, float* __restrict__ Aj)
{
    int i = blockIdx.x * blockDim.x + threadIdx.x;
    if (i >= 2 * B_ * 64 * GD_) return;
    const int n = i % GD_;
    const int p = (i / GD_) % 64;
    const int b = (i / (GD_ * 64)) % B_;
    const int which = i / (GD_ * 64 * B_);
    const float* xr = xf + (b * 64 + p) * 26;
    const float* wr = Wg1 + n * 180 + (which ? 26 : 0);
    float acc = 0.f;
    #pragma unroll
    for (int c = 0; c < 26; ++c) acc += xr[c] * wr[c];
    (which ? Aj : Ai)[((size_t)(b * 64 + p)) * GD_ + n] = acc;
}

// Q = qst @ Wg1[:, 52:].T + bg1
__global__ void proj_q_k(const float* __restrict__ qst, const float* __restrict__ Wg1,
                         const float* __restrict__ bg1, float* __restrict__ Q)
{
    int i = blockIdx.x * blockDim.x + threadIdx.x;
    if (i >= B_ * GD_) return;
    const int n = i % GD_;
    const int b = i / GD_;
    const float* wr = Wg1 + n * 180 + 52;
    const float* qr = qst + b * HID_;
    float acc = bg1[n];
    for (int c = 0; c < HID_; ++c) acc += qr[c] * wr[c];
    Q[i] = acc;
}

// f32 -> bf16 weight conversion (kept [n,k] row-major: that IS the WMMA B=W^T)
__global__ void to_bf16_k(const float* __restrict__ src, unsigned short* __restrict__ dst, int n)
{
    int i = blockIdx.x * blockDim.x + threadIdx.x;
    if (i < n) dst[i] = f2bf(src[i]);
}

// ---------------------------------------------------------------------------
// Fused pairwise g-MLP. Block = (b, k-object): 64 pair rows (all j) x 256.
// 8 wave32 waves; wave w owns N-tiles {2w, 2w+1} x all 4 M-tiles.
// Activations stay in LDS as bf16; 3x 256x256 layers via v_wmma bf16.
// ---------------------------------------------------------------------------
__device__ __forceinline__ void gemm_tiles(const unsigned short* __restrict__ Wbf,
                                           const unsigned short* __restrict__ hin,
                                           v8f acc[8], int w, int lane)
{
    const int half = lane >> 4;
    const int l16  = lane & 15;
    for (int k0 = 0; k0 < 256; k0 += 32) {
        Frag16 bfr[2];
        #pragma unroll
        for (int nt = 0; nt < 2; ++nt) {
            const int n = (2 * w + nt) * 16 + l16;
            // B-fragment: 16 contiguous bf16 along K of row n (W stored [n,k])
            const uint4* p = (const uint4*)(Wbf + n * 256 + k0 + 16 * half);
            bfr[nt].q[0] = p[0];
            bfr[nt].q[1] = p[1];
        }
        Frag16 afr[4];
        #pragma unroll
        for (int m = 0; m < 4; ++m) {
            const int row = m * 16 + l16;
            // A-fragment: 16 contiguous bf16 along K, K[base..base+15],
            // base = k0 + 16*half (matches B-fragment order)
            const unsigned short* hp = hin + row * 256 + k0 + 16 * half;
            afr[m].q[0] = *(const uint4*)(hp);
            afr[m].q[1] = *(const uint4*)(hp + 8);
        }
        #pragma unroll
        for (int m = 0; m < 4; ++m)
            #pragma unroll
            for (int nt = 0; nt < 2; ++nt)
                acc[m * 2 + nt] = __builtin_amdgcn_wmma_f32_16x16x32_bf16(
                    false, afr[m].v, false, bfr[nt].v,
                    (short)0, acc[m * 2 + nt], false, false);
    }
}

__device__ __forceinline__ void gemm_layer(const unsigned short* __restrict__ Wbf,
                                           const float* __restrict__ bias,
                                           const unsigned short* __restrict__ hin,
                                           unsigned short* __restrict__ hout)
{
    const int lane = threadIdx.x & 31;
    const int w    = threadIdx.x >> 5;
    const int half = lane >> 4;
    const int l16  = lane & 15;
    v8f acc[8];
    #pragma unroll
    for (int i = 0; i < 8; ++i)
        #pragma unroll
        for (int v = 0; v < 8; ++v) acc[i][v] = 0.0f;

    gemm_tiles(Wbf, hin, acc, w, lane);

    // C/D layout: lane<16 -> N=lane, M=vgpr ; lane>=16 -> N=lane-16, M=8+vgpr
    #pragma unroll
    for (int nt = 0; nt < 2; ++nt) {
        const int n = (2 * w + nt) * 16 + l16;
        const float bn = bias[n];
        #pragma unroll
        for (int m = 0; m < 4; ++m) {
            const v8f a = acc[m * 2 + nt];
            #pragma unroll
            for (int v = 0; v < 8; ++v) {
                const int row = m * 16 + 8 * half + v;
                hout[row * 256 + n] = f2bf(fmaxf(a[v] + bn, 0.0f));
            }
        }
    }
}

__device__ __forceinline__ void gemm_final(const unsigned short* __restrict__ Wbf,
                                           const float* __restrict__ bias,
                                           const unsigned short* __restrict__ hin,
                                           float* __restrict__ xg_b)
{
    const int lane = threadIdx.x & 31;
    const int w    = threadIdx.x >> 5;
    const int half = lane >> 4;
    const int l16  = lane & 15;
    v8f acc[8];
    #pragma unroll
    for (int i = 0; i < 8; ++i)
        #pragma unroll
        for (int v = 0; v < 8; ++v) acc[i][v] = 0.0f;

    gemm_tiles(Wbf, hin, acc, w, lane);

    #pragma unroll
    for (int nt = 0; nt < 2; ++nt) {
        const int n = (2 * w + nt) * 16 + l16;
        const float bn = bias[n];
        float s = 0.0f;
        #pragma unroll
        for (int m = 0; m < 4; ++m) {
            const v8f a = acc[m * 2 + nt];
            #pragma unroll
            for (int v = 0; v < 8; ++v) s += fmaxf(a[v] + bn, 0.0f);
        }
        s += __shfl_xor(s, 16, 32);            // combine rows of both lane halves
        if (half == 0) atomicAdd(&xg_b[n], s); // sum-pool over pairs
    }
}

__global__ void __launch_bounds__(256) gmlp_k(
    const float* __restrict__ Ai, const float* __restrict__ Aj, const float* __restrict__ Q,
    const unsigned short* __restrict__ W2, const float* __restrict__ b2,
    const unsigned short* __restrict__ W3, const float* __restrict__ b3,
    const unsigned short* __restrict__ W4, const float* __restrict__ b4,
    float* __restrict__ xg)
{
    __shared__ unsigned short hA[64 * 256];
    __shared__ unsigned short hB[64 * 256];
    const int b  = blockIdx.x >> 6;
    const int kk = blockIdx.x & 63;
    const float* AjR = Aj + ((size_t)(b * 64 + kk)) * GD_;
    const float* Qr  = Q + b * GD_;
    // h1 = relu(Ai[j] + Aj[k] + Q), bf16 in LDS
    for (int i = threadIdx.x; i < 64 * 256; i += 256) {
        const int j = i >> 8, n = i & 255;
        const float v = Ai[((size_t)(b * 64 + j)) * GD_ + n] + AjR[n] + Qr[n];
        hA[i] = f2bf(fmaxf(v, 0.0f));
    }
    __syncthreads();
    gemm_layer(W2, b2, hA, hB);
    __syncthreads();
    gemm_layer(W3, b3, hB, hA);
    __syncthreads();
    gemm_final(W4, b4, hA, xg + b * GD_);
}

// ---------------------------------------------------------------------------
// f-MLP + log_softmax: one block per batch element.
// ---------------------------------------------------------------------------
__global__ void __launch_bounds__(256) fmlp_k(
    const float* __restrict__ xg,
    const float* __restrict__ Wf1, const float* __restrict__ bf1,
    const float* __restrict__ Wf2, const float* __restrict__ bf2,
    const float* __restrict__ Wf3, const float* __restrict__ bf3v,
    float* __restrict__ out)
{
    __shared__ float s0[GD_], s1[GD_], s2[GD_], lg[AV_], red[2];
    const int b = blockIdx.x, t = threadIdx.x;
    s0[t] = xg[b * GD_ + t];
    __syncthreads();
    float acc = bf1[t];
    for (int c = 0; c < GD_; ++c) acc += s0[c] * Wf1[t * GD_ + c];
    s1[t] = fmaxf(acc, 0.0f);
    __syncthreads();
    acc = bf2[t];
    for (int c = 0; c < GD_; ++c) acc += s1[c] * Wf2[t * GD_ + c];
    s2[t] = fmaxf(acc, 0.0f);
    __syncthreads();
    if (t < AV_) {
        acc = bf3v[t];
        for (int c = 0; c < GD_; ++c) acc += s2[c] * Wf3[t * GD_ + c];
        lg[t] = acc;
    }
    __syncthreads();
    if (t == 0) {
        float m = -1e30f;
        for (int i = 0; i < AV_; ++i) m = fmaxf(m, lg[i]);
        float ss = 0.f;
        for (int i = 0; i < AV_; ++i) ss += expf(lg[i] - m);
        red[0] = m; red[1] = logf(ss);
    }
    __syncthreads();
    if (t < AV_) out[b * AV_ + t] = lg[t] - red[0] - red[1];
}

// ---------------------------------------------------------------------------
// Host launcher
// ---------------------------------------------------------------------------
extern "C" void kernel_launch(void* const* d_in, const int* in_sizes, int n_in,
                              void* d_out, int out_size, void* d_ws, size_t ws_size,
                              hipStream_t stream)
{
    const float* img   = (const float*)d_in[0];
    const int*   qidx  = (const int*)  d_in[1];
    const float* emb   = (const float*)d_in[2];
    const float* W_ih  = (const float*)d_in[3];
    const float* W_hh  = (const float*)d_in[4];
    const float* b_ih  = (const float*)d_in[5];
    const float* b_hh  = (const float*)d_in[6];
    const float* k1 = (const float*)d_in[7],  *cb1 = (const float*)d_in[8],
               * g1 = (const float*)d_in[9],  *be1 = (const float*)d_in[10];
    const float* k2 = (const float*)d_in[11], *cb2 = (const float*)d_in[12],
               * g2 = (const float*)d_in[13], *be2 = (const float*)d_in[14];
    const float* k3 = (const float*)d_in[15], *cb3 = (const float*)d_in[16],
               * g3 = (const float*)d_in[17], *be3 = (const float*)d_in[18];
    const float* k4 = (const float*)d_in[19], *cb4 = (const float*)d_in[20],
               * g4 = (const float*)d_in[21], *be4 = (const float*)d_in[22];
    const float* Wg1 = (const float*)d_in[23], *bg1 = (const float*)d_in[24];
    const float* Wg2 = (const float*)d_in[25], *bg2 = (const float*)d_in[26];
    const float* Wg3 = (const float*)d_in[27], *bg3 = (const float*)d_in[28];
    const float* Wg4 = (const float*)d_in[29], *bg4 = (const float*)d_in[30];
    const float* Wf1 = (const float*)d_in[31], *bf1 = (const float*)d_in[32];
    const float* Wf2 = (const float*)d_in[33], *bf2 = (const float*)d_in[34];
    const float* Wf3 = (const float*)d_in[35], *bf3 = (const float*)d_in[36];
    const float* h0  = (const float*)d_in[37], *c0  = (const float*)d_in[38];
    (void)in_sizes; (void)n_in; (void)out_size; (void)ws_size;

    char* wp = (char*)d_ws;
    size_t off = 0;
    auto take = [&](size_t bytes) -> void* {
        void* p = wp + off;
        off += (bytes + 255) & ~(size_t)255;
        return p;
    };
    float* y1   = (float*)take((size_t)B_ * 24 * 64 * 64 * 4);
    float* y2   = (float*)take((size_t)B_ * 24 * 32 * 32 * 4);
    float* y3   = (float*)take((size_t)B_ * 24 * 16 * 16 * 4);
    float* y4   = (float*)take((size_t)B_ * 24 * 8  * 8  * 4);
    float* st   = (float*)take(2 * 24 * 4);
    float* qst  = (float*)take((size_t)B_ * HID_ * 4);
    float* xf   = (float*)take((size_t)B_ * 64 * 26 * 4);
    float* Ai   = (float*)take((size_t)B_ * 64 * GD_ * 4);
    float* Aj   = (float*)take((size_t)B_ * 64 * GD_ * 4);
    float* Q    = (float*)take((size_t)B_ * GD_ * 4);
    unsigned short* W2b = (unsigned short*)take((size_t)GD_ * GD_ * 2);
    unsigned short* W3b = (unsigned short*)take((size_t)GD_ * GD_ * 2);
    unsigned short* W4b = (unsigned short*)take((size_t)GD_ * GD_ * 2);
    float* xg   = (float*)take((size_t)B_ * GD_ * 4);

    hipMemsetAsync(xg, 0, (size_t)B_ * GD_ * 4, stream);

    // Question encoder
    lstm_k<<<B_, 128, 0, stream>>>(qidx, emb, W_ih, W_hh, b_ih, b_hh, h0, c0, qst);

    // Conv stack (conv+relu, then train-mode BN stats + apply)
    auto conv_layer = [&](const float* x, const float* k, const float* cb,
                          const float* g, const float* be, float* y,
                          int Cin, int Hin, int Hout) {
        const int HW = Hout * Hout;
        const int total = B_ * 24 * HW;
        conv_relu_k<<<(total + 255) / 256, 256, 0, stream>>>(
            x, k, cb, y, Cin, 24, Hin, Hin, Hout, Hout);
        bn_stats_k<<<24, 256, 0, stream>>>(y, st, 24, HW);
        bn_apply_k<<<(total + 255) / 256, 256, 0, stream>>>(y, st, g, be, 24, HW);
    };
    conv_layer(img, k1, cb1, g1, be1, y1, 3, 128, 64);
    conv_layer(y1,  k2, cb2, g2, be2, y2, 24, 64, 32);
    conv_layer(y2,  k3, cb3, g3, be3, y3, 24, 32, 16);
    conv_layer(y3,  k4, cb4, g4, be4, y4, 24, 16, 8);

    // Object features + first-layer decomposition
    build_xf_k<<<(B_ * 64 * 26 + 255) / 256, 256, 0, stream>>>(y4, xf);
    proj_a_k<<<(2 * B_ * 64 * GD_ + 255) / 256, 256, 0, stream>>>(xf, Wg1, Ai, Aj);
    proj_q_k<<<(B_ * GD_ + 255) / 256, 256, 0, stream>>>(qst, Wg1, bg1, Q);

    // bf16 weight prep for WMMA
    to_bf16_k<<<(GD_ * GD_ + 255) / 256, 256, 0, stream>>>(Wg2, W2b, GD_ * GD_);
    to_bf16_k<<<(GD_ * GD_ + 255) / 256, 256, 0, stream>>>(Wg3, W3b, GD_ * GD_);
    to_bf16_k<<<(GD_ * GD_ + 255) / 256, 256, 0, stream>>>(Wg4, W4b, GD_ * GD_);

    // Fused pairwise g-MLP: one block per (b, k-object)
    gmlp_k<<<B_ * 64, 256, 0, stream>>>(Ai, Aj, Q, W2b, bg2, W3b, bg3, W4b, bg4, xg);

    // f-MLP + log_softmax
    fmlp_k<<<B_, 256, 0, stream>>>(xg, Wf1, bf1, Wf2, bf2, Wf3, bf3, (float*)d_out);
}
